// ZR2SBlock2D_62328565400318
// MI455X (gfx1250) — compile-verified, matrix operand order checked by reading, not run on hardware
//
#include <hip/hip_runtime.h>
#include <stdint.h>

// ---------------- problem constants ----------------
#define B_  8
#define H_  48
#define W_  48
#define C_  192
#define D_  384           // d_inner
#define NS  16            // state size
#define RK  12            // dt rank
#define KS  2             // scan directions
#define L_  (H_ * W_)     // 2304
#define T_  (B_ * L_)     // 18432 tokens
#define E_  44            // dt_rank + 2*N
#define EP  48            // E_ padded to multiple of 16
#define KDT 32            // dt-proj K (12) padded to 32
#define EPSV 1e-6f

typedef __attribute__((ext_vector_type(16))) __bf16 v16bf;
typedef __attribute__((ext_vector_type(8)))  float  v8f;

static __device__ __forceinline__ __bf16 tobf(float f) {
  union { float f; uint32_t u; } c; c.f = f;
  uint32_t b = c.u;
  uint32_t r = (b + 0x7FFFu + ((b >> 16) & 1u)) >> 16;   // RNE truncate
  union { unsigned short s; __bf16 h; } o; o.s = (unsigned short)r;
  return o.h;
}

// =====================================================================
// Weight prep 1: padded fp32 transposes.
//  wxp_t[k][d][e(48)] = x_proj_w[k][e][d], zero for e>=44
//  dtw_t[k][r(32)][d] = dt_proj_w[k][d][r], zero for r>=12
// =====================================================================
__global__ void prep_weights_kernel(const float* __restrict__ xpw,
                                    const float* __restrict__ dtw,
                                    float* __restrict__ wxp_t,
                                    float* __restrict__ dtw_t) {
  int i = blockIdx.x * blockDim.x + threadIdx.x;
  if (i < KS * D_ * EP) {
    int k = i / (D_ * EP);
    int rem = i - k * D_ * EP;
    int d = rem / EP;
    int e = rem - d * EP;
    wxp_t[i] = (e < E_) ? xpw[(size_t)k * E_ * D_ + (size_t)e * D_ + d] : 0.f;
  }
  if (i < KS * KDT * D_) {
    int k = i / (KDT * D_);
    int rem = i - k * KDT * D_;
    int r = rem / D_;
    int d = rem - r * D_;
    dtw_t[i] = (r < RK) ? dtw[(size_t)k * D_ * RK + (size_t)d * RK + r] : 0.f;
  }
}

// =====================================================================
// Weight prep 2: pack fp32 W[K,N] (row-major, ldw=N, K%32==0, N%16==0)
// into per-lane WMMA B-fragment order:
//   out[((kb*Ntiles + nt)*32 + lane)*16 + {2j,2j+1}] = W[kp,col],W[kp+1,col]
//   kp = kb*32 + (j>=4)*16 + (lane>>4)*8 + (j&3)*2 ; col = nt*16 + (lane&15)
// One thread per (kb, nt, lane) -> writes 16 contiguous bf16 (32B).
// =====================================================================
__global__ void pack_w_kernel(const float* __restrict__ W, int ldw, int Ntiles,
                              int Kblocks, __bf16* __restrict__ out) {
  int tid = blockIdx.x * blockDim.x + threadIdx.x;
  if (tid >= Kblocks * Ntiles * 32) return;
  int lane = tid & 31;
  int nt = (tid >> 5) % Ntiles;
  int kb = (tid >> 5) / Ntiles;
  int half = lane >> 4;
  int col = nt * 16 + (lane & 15);
  __bf16* o = out + (size_t)tid * 16;
#pragma unroll
  for (int j = 0; j < 8; ++j) {
    int kp = (kb << 5) + ((j >> 2) << 4) + (half << 3) + ((j & 3) << 1);
    o[2 * j]     = tobf(W[(size_t)kp * ldw + col]);
    o[2 * j + 1] = tobf(W[(size_t)(kp + 1) * ldw + col]);
  }
}

// =====================================================================
// Pre-LN over C, one wave32 per token; writes bf16 (GEMM A operand).
// =====================================================================
__global__ void ln_in_kernel(const float* __restrict__ x,
                             const float* __restrict__ g,
                             const float* __restrict__ b,
                             __bf16* __restrict__ xnb) {
  int wave = threadIdx.x >> 5, lane = threadIdx.x & 31;
  int t = blockIdx.x * 8 + wave;
  const float* xr = x + (size_t)t * C_;
  float v[6], s = 0.f, s2 = 0.f;
#pragma unroll
  for (int i = 0; i < 6; ++i) { v[i] = xr[lane + 32 * i]; s += v[i]; s2 += v[i] * v[i]; }
#pragma unroll
  for (int o = 16; o > 0; o >>= 1) { s += __shfl_xor(s, o); s2 += __shfl_xor(s2, o); }
  float mean = s * (1.f / C_);
  float inv  = rsqrtf(s2 * (1.f / C_) - mean * mean + EPSV);
  __bf16* xo = xnb + (size_t)t * C_;
#pragma unroll
  for (int i = 0; i < 6; ++i) {
    int c = lane + 32 * i;
    xo[c] = tobf((v[i] - mean) * inv * g[c] + b[c]);
  }
}

// =====================================================================
// WMMA GEMM: O[M, Ntiles*16] = A[M, Kp](bf16, lda==Kp) @ Wp(packed bf16)
// All dims padded: no bounds checks in hot loop.
// block = 128 (4 waves): same 16-row A panel staged once in LDS,
// each wave owns one 16x16 N tile. Inner loop: 2 ds_b128 + 2 global_b128
// + 1 v_wmma_f32_16x16x32_bf16.
// =====================================================================
__global__ void gemm_bf16_wmma(const __bf16* __restrict__ Ab,
                               const __bf16* __restrict__ Wp,
                               float* __restrict__ O, int ldo,
                               int Kp, int Ntiles) {
  const int lane = threadIdx.x & 31;
  const int wave = threadIdx.x >> 5;
  const int half = lane >> 4;
  const int mn   = lane & 15;
  const int mBase = blockIdx.y * 16;
  const int ntile = blockIdx.x * 4 + wave;

  __shared__ __align__(16) __bf16 sA[16 * 384];
  {
    const uint4* src = (const uint4*)(Ab + (size_t)mBase * Kp);
    uint4* dst = (uint4*)sA;
    int chunks = (16 * Kp) >> 3;                 // 8 bf16 per uint4
    for (int i = threadIdx.x; i < chunks; i += 128) dst[i] = src[i];
  }
  __syncthreads();
  if (ntile >= Ntiles) return;                   // wave-uniform

  v8f acc = {};
  const __bf16* sArow = sA + mn * Kp + half * 8;
  const __bf16* wp = Wp + ((size_t)ntile * 32 + lane) * 16;
  const size_t wstep = (size_t)Ntiles * 32 * 16; // per K-block advance
  const int Kblocks = Kp >> 5;
  for (int kb = 0; kb < Kblocks; ++kb) {
    union { v16bf v; uint4 q[2]; } a, b;
    a.q[0] = *(const uint4*)(sArow + (kb << 5));
    a.q[1] = *(const uint4*)(sArow + (kb << 5) + 16);
    b.q[0] = *(const uint4*)(wp);
    b.q[1] = *(const uint4*)(wp + 8);
    wp += wstep;
    acc = __builtin_amdgcn_wmma_f32_16x16x32_bf16(false, a.v, false, b.v,
                                                  (short)0, acc, false, false);
  }
#pragma unroll
  for (int i = 0; i < 8; ++i) {                  // C/D: VGPR i -> M = i + 8*half
    int r = mBase + i + half * 8;
    O[(size_t)r * ldo + ntile * 16 + mn] = acc[i];
  }
}

// =====================================================================
// Depthwise 3x3 conv + bias + SiLU. Writes fp32 (scan input u) and bf16
// (x_proj GEMM A operand).
// =====================================================================
__global__ void conv_silu_kernel(const float* __restrict__ xz,
                                 const float* __restrict__ cw,
                                 const float* __restrict__ cb,
                                 float* __restrict__ xt,
                                 __bf16* __restrict__ xtb) {
  int d = threadIdx.x;            // 0..383
  int t = blockIdx.x;             // token
  int b = t / L_;
  int l = t - b * L_;
  int h = l / W_;
  int w = l - h * W_;
  float acc = cb[d];
#pragma unroll
  for (int kh = 0; kh < 3; ++kh) {
    int hh = h + kh - 1;
    if (hh < 0 || hh >= H_) continue;
#pragma unroll
    for (int kw = 0; kw < 3; ++kw) {
      int ww = w + kw - 1;
      if (ww < 0 || ww >= W_) continue;
      size_t idx = ((size_t)(b * L_ + hh * W_ + ww)) * (2 * D_) + d;
      acc += xz[idx] * cw[d * 9 + kh * 3 + kw];
    }
  }
  float sv = acc / (1.f + __expf(-acc));
  xt[(size_t)t * D_ + d]  = sv;
  xtb[(size_t)t * D_ + d] = tobf(sv);
}

// =====================================================================
// xdbl[:, 0:32] fp32 -> bf16 (dt-proj GEMM A operand; K padded to 32,
// cols 12..31 hit zero weight rows so their values are don't-care).
// =====================================================================
__global__ void cvt_xdbl_kernel(const float* __restrict__ xdbl,
                                __bf16* __restrict__ xdblb, int total) {
  int i = blockIdx.x * blockDim.x + threadIdx.x;
  if (i >= total) return;
  int c = i & (KDT - 1);
  int t = i >> 5;                 // (k*T + token)
  xdblb[i] = tobf(xdbl[(size_t)t * EP + c]);
}

// ---------------- delta = softplus(dts + dt_bias), in place ----------------
__global__ void softplus_kernel(float* __restrict__ dts,
                                const float* __restrict__ bias, int total) {
  int i = blockIdx.x * blockDim.x + threadIdx.x;
  if (i >= total) return;
  int d = i % D_;
  int k = i / (T_ * D_);
  float xv = dts[i] + bias[k * D_ + d];
  dts[i] = (xv > 20.f) ? xv : log1pf(__expf(xv));
}

// =====================================================================
// Selective scan: thread per (b,k,d); h[16], A[16] in registers, B/C of
// the step broadcast through LDS. k=1 walks tokens in reverse and writes
// at token position -> already un-reversed.
// =====================================================================
__global__ void scan_kernel(const float* __restrict__ xt,
                            const float* __restrict__ dts,
                            const float* __restrict__ xdbl,
                            const float* __restrict__ A_log,
                            const float* __restrict__ Ds,
                            float* __restrict__ ys0,
                            float* __restrict__ ys1) {
  int bk = blockIdx.x;
  int b = bk >> 1;
  int k = bk & 1;
  int d = blockIdx.y * 128 + threadIdx.x;
  __shared__ float sBC[32];            // [0:16)=B_t, [16:32)=C_t
  float a[NS], h[NS];
#pragma unroll
  for (int n = 0; n < NS; ++n) {
    a[n] = -__expf(A_log[((size_t)k * D_ + d) * NS + n]);
    h[n] = 0.f;
  }
  float dsv = Ds[k * D_ + d];
  float* ysk = k ? ys1 : ys0;
  for (int t = 0; t < L_; ++t) {
    int l = k ? (L_ - 1 - t) : t;
    size_t tkn = (size_t)b * L_ + l;
    __syncthreads();
    if (threadIdx.x < 32)
      sBC[threadIdx.x] = xdbl[((size_t)k * T_ + tkn) * EP + RK + threadIdx.x];
    __syncthreads();
    float delta = dts[((size_t)k * T_ + tkn) * D_ + d];
    float u = xt[tkn * D_ + d];
    float du = delta * u;
    float y = 0.f;
#pragma unroll
    for (int n = 0; n < NS; ++n) {
      float dA = __expf(delta * a[n]);
      h[n] = h[n] * dA + du * sBC[n];
      y += h[n] * sBC[16 + n];
    }
    ysk[tkn * D_ + d] = y + dsv * u;
  }
}

// =====================================================================
// Merge scans + out-LN over D + silu(z) gating; writes bf16 (out_proj A).
// =====================================================================
__global__ void merge_ln_gate_kernel(const float* __restrict__ ys0,
                                     const float* __restrict__ ys1,
                                     const float* __restrict__ xz,
                                     const float* __restrict__ g,
                                     const float* __restrict__ bb,
                                     __bf16* __restrict__ ygb) {
  int t = blockIdx.x;
  int d = threadIdx.x;                 // 0..383
  int lane = d & 31, wid = d >> 5;
  float v = ys0[(size_t)t * D_ + d] + ys1[(size_t)t * D_ + d];
  float s = v, s2 = v * v;
#pragma unroll
  for (int o = 16; o > 0; o >>= 1) { s += __shfl_xor(s, o); s2 += __shfl_xor(s2, o); }
  __shared__ float r1[12], r2[12], mv[2];
  if (lane == 0) { r1[wid] = s; r2[wid] = s2; }
  __syncthreads();
  if (d == 0) {
    float ts = 0.f, ts2 = 0.f;
    for (int i = 0; i < 12; ++i) { ts += r1[i]; ts2 += r2[i]; }
    float mean = ts * (1.f / D_);
    mv[0] = mean;
    mv[1] = rsqrtf(ts2 * (1.f / D_) - mean * mean + EPSV);
  }
  __syncthreads();
  float ln = (v - mv[0]) * mv[1] * g[d] + bb[d];
  float z = xz[(size_t)t * (2 * D_) + D_ + d];
  ygb[(size_t)t * D_ + d] = tobf(ln * (z / (1.f + __expf(-z))));
}

// ---------------- residual add ----------------
__global__ void residual_kernel(const float* __restrict__ x,
                                const float* __restrict__ proj,
                                float* __restrict__ out, int total) {
  int i = blockIdx.x * blockDim.x + threadIdx.x;
  if (i < total) out[i] = x[i] + proj[i];
}

// =====================================================================
// host orchestration
// =====================================================================
extern "C" void kernel_launch(void* const* d_in, const int* in_sizes, int n_in,
                              void* d_out, int out_size, void* d_ws, size_t ws_size,
                              hipStream_t stream) {
  (void)in_sizes; (void)n_in; (void)out_size; (void)ws_size;

  const float* x         = (const float*)d_in[0];
  const float* norm_g    = (const float*)d_in[1];
  const float* norm_b    = (const float*)d_in[2];
  const float* in_proj_w = (const float*)d_in[3];   // [192,768]
  const float* conv_w    = (const float*)d_in[4];   // [384,1,3,3]
  const float* conv_b    = (const float*)d_in[5];
  const float* x_proj_w  = (const float*)d_in[6];   // [2,44,384]
  const float* dt_proj_w = (const float*)d_in[7];   // [2,384,12]
  const float* dt_proj_b = (const float*)d_in[8];
  const float* A_log     = (const float*)d_in[9];
  const float* Ds        = (const float*)d_in[10];
  const float* out_ng    = (const float*)d_in[11];
  const float* out_nb    = (const float*)d_in[12];
  const float* out_proj_w= (const float*)d_in[13];  // [384,192]
  float* out = (float*)d_out;

  // byte-cursor workspace allocator (256B aligned)
  char* cur = (char*)d_ws;
  auto alloc = [&cur](size_t bytes) {
    char* p = cur;
    cur += (bytes + 255) & ~(size_t)255;
    return (void*)p;
  };
  float*  xz    = (float*)alloc((size_t)T_ * 2 * D_ * 4);
  float*  xt    = (float*)alloc((size_t)T_ * D_ * 4);
  float*  xdbl  = (float*)alloc((size_t)KS * T_ * EP * 4);
  float*  dts   = (float*)alloc((size_t)KS * T_ * D_ * 4);
  float*  ys0   = (float*)alloc((size_t)T_ * D_ * 4);
  float*  ys1   = (float*)alloc((size_t)T_ * D_ * 4);
  float*  proj  = (float*)alloc((size_t)T_ * C_ * 4);
  float*  wxp_t = (float*)alloc((size_t)KS * D_ * EP * 4);
  float*  dtw_t = (float*)alloc((size_t)KS * KDT * D_ * 4);
  __bf16* xnb   = (__bf16*)alloc((size_t)T_ * C_ * 2);
  __bf16* xtb   = (__bf16*)alloc((size_t)T_ * D_ * 2);
  __bf16* xdblb = (__bf16*)alloc((size_t)KS * T_ * KDT * 2);
  __bf16* ygb   = (__bf16*)alloc((size_t)T_ * D_ * 2);
  __bf16* p_in  = (__bf16*)alloc((size_t)6 * 48 * 512 * 2);   // K=192,N=768
  __bf16* p_xp0 = (__bf16*)alloc((size_t)12 * 3 * 512 * 2);   // K=384,N=48
  __bf16* p_xp1 = (__bf16*)alloc((size_t)12 * 3 * 512 * 2);
  __bf16* p_dt0 = (__bf16*)alloc((size_t)1 * 24 * 512 * 2);   // K=32, N=384
  __bf16* p_dt1 = (__bf16*)alloc((size_t)1 * 24 * 512 * 2);
  __bf16* p_out = (__bf16*)alloc((size_t)12 * 12 * 512 * 2);  // K=384,N=192

  // 1. padded weight transposes + fragment packing
  prep_weights_kernel<<<(KS * D_ * EP + 255) / 256, 256, 0, stream>>>(
      x_proj_w, dt_proj_w, wxp_t, dtw_t);
  pack_w_kernel<<<(6 * 48 * 32 + 255) / 256, 256, 0, stream>>>(
      in_proj_w, 2 * D_, 48, 6, p_in);
  pack_w_kernel<<<(12 * 3 * 32 + 255) / 256, 256, 0, stream>>>(
      wxp_t, EP, 3, 12, p_xp0);
  pack_w_kernel<<<(12 * 3 * 32 + 255) / 256, 256, 0, stream>>>(
      wxp_t + (size_t)D_ * EP, EP, 3, 12, p_xp1);
  pack_w_kernel<<<(1 * 24 * 32 + 255) / 256, 256, 0, stream>>>(
      dtw_t, D_, 24, 1, p_dt0);
  pack_w_kernel<<<(1 * 24 * 32 + 255) / 256, 256, 0, stream>>>(
      dtw_t + (size_t)KDT * D_, D_, 24, 1, p_dt1);
  pack_w_kernel<<<(12 * 12 * 32 + 255) / 256, 256, 0, stream>>>(
      out_proj_w, C_, 12, 12, p_out);

  // 2. pre-LN (fp32 in, bf16 out)
  ln_in_kernel<<<T_ / 8, 256, 0, stream>>>(x, norm_g, norm_b, xnb);

  // 3. in_proj: xz[T,768] = xnb[T,192] @ p_in
  gemm_bf16_wmma<<<dim3(48 / 4, T_ / 16), 128, 0, stream>>>(
      xnb, p_in, xz, 2 * D_, C_, 48);

  // 4. depthwise conv + SiLU
  conv_silu_kernel<<<T_, D_, 0, stream>>>(xz, conv_w, conv_b, xt, xtb);

  // 5. x_proj per direction: xdbl_k[T,48] = xtb[T,384] @ p_xp{k}
  gemm_bf16_wmma<<<dim3(1, T_ / 16), 128, 0, stream>>>(
      xtb, p_xp0, xdbl, EP, D_, 3);
  gemm_bf16_wmma<<<dim3(1, T_ / 16), 128, 0, stream>>>(
      xtb, p_xp1, xdbl + (size_t)T_ * EP, EP, D_, 3);

  // 6. dt_proj: convert xdbl[:, :32] to bf16, then dts_k = xdblb_k @ p_dt{k}
  cvt_xdbl_kernel<<<(KS * T_ * KDT + 255) / 256, 256, 0, stream>>>(
      xdbl, xdblb, KS * T_ * KDT);
  gemm_bf16_wmma<<<dim3(24 / 4, T_ / 16), 128, 0, stream>>>(
      xdblb, p_dt0, dts, D_, KDT, 24);
  gemm_bf16_wmma<<<dim3(24 / 4, T_ / 16), 128, 0, stream>>>(
      xdblb + (size_t)T_ * KDT, p_dt1, dts + (size_t)T_ * D_, D_, KDT, 24);

  // 7. delta = softplus(dts + bias)
  int tot = KS * T_ * D_;
  softplus_kernel<<<(tot + 255) / 256, 256, 0, stream>>>(dts, dt_proj_b, tot);

  // 8. selective scan, both directions
  scan_kernel<<<dim3(B_ * KS, D_ / 128), 128, 0, stream>>>(
      xt, dts, xdbl, A_log, Ds, ys0, ys1);

  // 9. merge + out-LN + gating (bf16 out)
  merge_ln_gate_kernel<<<T_, D_, 0, stream>>>(ys0, ys1, xz, out_ng, out_nb, ygb);

  // 10. out_proj: proj[T,192] = ygb[T,384] @ p_out
  gemm_bf16_wmma<<<dim3(12 / 4, T_ / 16), 128, 0, stream>>>(
      ygb, p_out, proj, C_, D_, 12);

  // 11. residual
  residual_kernel<<<(T_ * C_ + 255) / 256, 256, 0, stream>>>(x, proj, out, T_ * C_);
}
